// PointNetMSG_53171695125110
// MI455X (gfx1250) — matrix operand order repaired and checked by reading, not compiled, not run on hardware
//
#include <hip/hip_runtime.h>
#include <hip/hip_bf16.h>

typedef __attribute__((ext_vector_type(16))) _Float16 v16h;
typedef __attribute__((ext_vector_type(8)))  float    v8f;

#define BATCH 8
#define NPTS  8192
#define NSAMP 1024
#define BN_EPS 1e-5f

// ---------------------------------------------------------------------------
// Farthest point sampling: 1 block per batch, 256 threads, dist in registers.
// ---------------------------------------------------------------------------
__global__ __launch_bounds__(256) void fps_kernel(const float* __restrict__ xyz,
                                                  float* __restrict__ new_xyz,
                                                  int* __restrict__ fps_idx) {
    __shared__ float rv[256];
    __shared__ int   ri[256];
    const int b   = blockIdx.x;
    const int tid = threadIdx.x;
    const float* xb = xyz + (size_t)b * NPTS * 3;

    float dist[32];
#pragma unroll
    for (int i = 0; i < 32; ++i) dist[i] = 1e10f;

    int far = 0;
    for (int it = 0; it < NSAMP; ++it) {
        const float cx = xb[far * 3 + 0];
        const float cy = xb[far * 3 + 1];
        const float cz = xb[far * 3 + 2];
        if (tid == 0) {
            fps_idx[b * NSAMP + it] = far;
            float* o = new_xyz + (size_t)(b * NSAMP + it) * 3;
            o[0] = cx; o[1] = cy; o[2] = cz;
        }
        float bv = -1.0f; int bi = 0;
#pragma unroll 4
        for (int i = 0; i < 32; ++i) {
            const int n = tid + (i << 8);
            const float dx = xb[n * 3 + 0] - cx;
            const float dy = xb[n * 3 + 1] - cy;
            const float dz = xb[n * 3 + 2] - cz;
            const float d  = dx * dx + dy * dy + dz * dz;
            const float nd = fminf(dist[i], d);
            dist[i] = nd;
            if (nd > bv) { bv = nd; bi = n; }
        }
        rv[tid] = bv; ri[tid] = bi;
        __syncthreads();
        for (int off = 128; off > 0; off >>= 1) {
            if (tid < off) {
                const float v2 = rv[tid + off];
                const int   i2 = ri[tid + off];
                if (v2 > rv[tid] || (v2 == rv[tid] && i2 < ri[tid])) { rv[tid] = v2; ri[tid] = i2; }
            }
            __syncthreads();
        }
        far = ri[0];
        __syncthreads();
    }
}

// ---------------------------------------------------------------------------
// Ball query + gather: one wave32 per query point; first-K-by-index semantics.
// Rows are 32-ch padded f16, stored with 4x b128 stores.
// ---------------------------------------------------------------------------
__global__ __launch_bounds__(128) void ball_gather_kernel(const float* __restrict__ xyz,
                                                          const float* __restrict__ feat,
                                                          const float* __restrict__ new_xyz,
                                                          _Float16* __restrict__ Xout,
                                                          int K, float r2) {
    const int lane = threadIdx.x & 31;
    const int q = blockIdx.x * (blockDim.x >> 5) + (threadIdx.x >> 5);
    if (q >= BATCH * NSAMP) return;
    const int b = q / NSAMP;
    const float* xb = xyz  + (size_t)b * NPTS * 3;
    const float* fb = feat + (size_t)b * NPTS * 3;

    const float nx = new_xyz[q * 3 + 0];
    const float ny = new_xyz[q * 3 + 1];
    const float nz = new_xyz[q * 3 + 2];
    const float s2 = nx * nx + ny * ny + nz * nz;

    int count = 0;
    int first_n = -1;
#pragma unroll 1
    for (int base = 0; base < NPTS && count < K; base += 32) {
        const int n = base + lane;
        const float px = xb[n * 3 + 0];
        const float py = xb[n * 3 + 1];
        const float pz = xb[n * 3 + 2];
        const float p2  = px * px + py * py + pz * pz;
        const float dot = nx * px + ny * py + nz * pz;
        const float sqr = s2 + p2 - 2.0f * dot;   // identical formula to reference
        const bool inr = (sqr <= r2);
        const unsigned mm = (unsigned)__ballot(inr);
        const int pos = count + __popc(mm & ((1u << lane) - 1u));
        if (inr && pos < K) {
            union { _Float16 h[32]; uint4 v[4]; } t;
#pragma unroll
            for (int c = 0; c < 32; ++c) t.h[c] = (_Float16)0.0f;
            t.h[0] = (_Float16)fb[n * 3 + 0];
            t.h[1] = (_Float16)fb[n * 3 + 1];
            t.h[2] = (_Float16)fb[n * 3 + 2];
            t.h[3] = (_Float16)(px - nx);
            t.h[4] = (_Float16)(py - ny);
            t.h[5] = (_Float16)(pz - nz);
            uint4* dst = (uint4*)(Xout + ((size_t)q * K + pos) * 32);
            dst[0] = t.v[0]; dst[1] = t.v[1]; dst[2] = t.v[2]; dst[3] = t.v[3];
        }
        if (first_n < 0 && mm != 0u) first_n = base + __ffs(mm) - 1;
        count += __popc(mm);
    }
    if (count < K) {
        const int n = (first_n >= 0) ? first_n : (NPTS - 1);
        const float px = xb[n * 3 + 0], py = xb[n * 3 + 1], pz = xb[n * 3 + 2];
        union { _Float16 h[32]; uint4 v[4]; } t;
#pragma unroll
        for (int c = 0; c < 32; ++c) t.h[c] = (_Float16)0.0f;
        t.h[0] = (_Float16)fb[n * 3 + 0];
        t.h[1] = (_Float16)fb[n * 3 + 1];
        t.h[2] = (_Float16)fb[n * 3 + 2];
        t.h[3] = (_Float16)(px - nx);
        t.h[4] = (_Float16)(py - ny);
        t.h[5] = (_Float16)(pz - nz);
        for (int rr = count + lane; rr < K; rr += 32) {
            uint4* dst = (uint4*)(Xout + ((size_t)q * K + rr) * 32);
            dst[0] = t.v[0]; dst[1] = t.v[1]; dst[2] = t.v[2]; dst[3] = t.v[3];
        }
    }
}

// ---------------------------------------------------------------------------
// Weight prep: W[cin x cout] f32  ->  Wt[cout x cin_pad] f16, zero-padded.
// ---------------------------------------------------------------------------
__global__ __launch_bounds__(256) void prep_w_kernel(const float* __restrict__ W,
                                                     _Float16* __restrict__ Wt,
                                                     int cin, int cin_pad, int cout) {
    const int total = cout * cin_pad;
    for (int idx = blockIdx.x * blockDim.x + threadIdx.x; idx < total;
         idx += gridDim.x * blockDim.x) {
        const int n = idx / cin_pad;
        const int k = idx % cin_pad;
        Wt[idx] = (k < cin) ? (_Float16)W[(size_t)k * cout + n] : (_Float16)0.0f;
    }
}

// ---------------------------------------------------------------------------
// WMMA GEMM: one wave computes a 16x32 output tile. K-loop unrolled by 2 with
// true double buffering (each phase owns its fragment registers -> no
// register-rotation movs). Y f16 + per-channel BN stats via f32 atomics.
// ---------------------------------------------------------------------------
__device__ __forceinline__ v16h load_afrag(const _Float16* __restrict__ Xrow,
                                           int kb, int half) {
    v16h a;
#pragma unroll
    for (int j = 0; j < 8; ++j)  a[j] = Xrow[kb + j + 8 * half];
#pragma unroll
    for (int j = 8; j < 16; ++j) a[j] = Xrow[kb + j + 8 + 8 * half];
    return a;
}
__device__ __forceinline__ v16h load_bfrag(const _Float16* __restrict__ Wrow,
                                           int kb, int half) {
    v16h bm;
#pragma unroll
    for (int j = 0; j < 8; ++j)  bm[j] = Wrow[kb + j + 8 * half];
#pragma unroll
    for (int j = 8; j < 16; ++j) bm[j] = Wrow[kb + j + 8 + 8 * half];
    return bm;
}

__global__ __launch_bounds__(256) void gemm_kernel(const _Float16* __restrict__ X,
                                                   const _Float16* __restrict__ Wt,
                                                   const float* __restrict__ bias,
                                                   _Float16* __restrict__ Y,
                                                   float* __restrict__ stats,
                                                   long long M, int cin_pad, int cout) {
    const int lane = threadIdx.x & 31;
    const int l16  = lane & 15;
    const int half = lane >> 4;
    const long long wave   = (long long)blockIdx.x * (blockDim.x >> 5) + (threadIdx.x >> 5);
    const int       ntile2 = cout >> 5;                 // 16x32 tiles along N
    const long long mtile  = M >> 4;
    if (wave >= mtile * (long long)ntile2) return;
    const long long m0  = (wave / ntile2) << 4;
    const int       n0  = (int)(wave % ntile2) << 5;
    const int       nc0 = n0 + l16;
    const int       nc1 = n0 + 16 + l16;

    const _Float16* Xrow  = X  + (m0 + l16) * (long long)cin_pad;
    const _Float16* Wrow0 = Wt + (long long)nc0 * cin_pad;
    const _Float16* Wrow1 = Wt + (long long)nc1 * cin_pad;

    v8f acc0 = {};
    v8f acc1 = {};
    // phase-0 buffers (stable registers across iterations)
    v16h a0 = load_afrag(Xrow, 0, half);
    v16h p0 = load_bfrag(Wrow0, 0, half);
    v16h q0 = load_bfrag(Wrow1, 0, half);
#pragma unroll 1
    for (int kb = 0; kb < cin_pad; kb += 64) {
        v16h a1, p1, q1;
        const bool has1 = (kb + 32) < cin_pad;
        if (has1) {                                    // phase-1 fragments
            a1 = load_afrag(Xrow, kb + 32, half);
            p1 = load_bfrag(Wrow0, kb + 32, half);
            q1 = load_bfrag(Wrow1, kb + 32, half);
        }
        acc0 = __builtin_amdgcn_wmma_f32_16x16x32_f16(false, a0, false, p0,
                                                      (short)0, acc0, false, false);
        acc1 = __builtin_amdgcn_wmma_f32_16x16x32_f16(false, a0, false, q0,
                                                      (short)0, acc1, false, false);
        if (kb + 64 < cin_pad) {                       // refill phase-0 for next iter
            __builtin_prefetch(Xrow + kb + 96, 0, 3);
            a0 = load_afrag(Xrow, kb + 64, half);
            p0 = load_bfrag(Wrow0, kb + 64, half);
            q0 = load_bfrag(Wrow1, kb + 64, half);
        }
        if (has1) {
            acc0 = __builtin_amdgcn_wmma_f32_16x16x32_f16(false, a1, false, p1,
                                                          (short)0, acc0, false, false);
            acc1 = __builtin_amdgcn_wmma_f32_16x16x32_f16(false, a1, false, q1,
                                                          (short)0, acc1, false, false);
        }
    }

    const float bv0 = bias[nc0];
    const float bv1 = bias[nc1];
    float sum0 = 0.f, ssq0 = 0.f, sum1 = 0.f, ssq1 = 0.f;
#pragma unroll
    for (int i = 0; i < 8; ++i) {
        const long long row = (m0 + i + 8 * half) * (long long)cout;
        const float v0 = acc0[i] + bv0;
        const float v1 = acc1[i] + bv1;
        sum0 += v0; ssq0 += v0 * v0;
        sum1 += v1; ssq1 += v1 * v1;
        Y[row + nc0] = (_Float16)v0;
        Y[row + nc1] = (_Float16)v1;
    }
    sum0 += __shfl_xor(sum0, 16, 32);  ssq0 += __shfl_xor(ssq0, 16, 32);
    sum1 += __shfl_xor(sum1, 16, 32);  ssq1 += __shfl_xor(ssq1, 16, 32);
    if (half == 0) {
        atomicAdd(&stats[2 * nc0 + 0], sum0);
        atomicAdd(&stats[2 * nc0 + 1], ssq0);
        atomicAdd(&stats[2 * nc1 + 0], sum1);
        atomicAdd(&stats[2 * nc1 + 1], ssq1);
    }
}

// ---------------------------------------------------------------------------
// BatchNorm (training stats) + ReLU, in place on f16; optional f32 output.
// ---------------------------------------------------------------------------
__global__ __launch_bounds__(256) void bn_relu_kernel(_Float16* __restrict__ Y,
                                                      const float* __restrict__ stats,
                                                      const float* __restrict__ gamma,
                                                      const float* __restrict__ beta,
                                                      long long M, int cout,
                                                      float* __restrict__ out32) {
    const long long total = M * (long long)cout;
    const float invM = 1.0f / (float)M;
    for (long long idx = (long long)blockIdx.x * blockDim.x + threadIdx.x; idx < total;
         idx += (long long)gridDim.x * blockDim.x) {
        const int c = (int)(idx % cout);
        const float mean = stats[2 * c + 0] * invM;
        const float var  = stats[2 * c + 1] * invM - mean * mean;
        const float sc = gamma[c] * rsqrtf(var + BN_EPS);
        const float sh = beta[c] - mean * sc;
        float v = (float)Y[idx] * sc + sh;
        v = fmaxf(v, 0.0f);
        Y[idx] = (_Float16)v;
        if (out32) out32[idx] = v;
    }
}

// ---------------------------------------------------------------------------
// Max pool over K samples per query point into concat buffer (320 ch).
// ---------------------------------------------------------------------------
__global__ __launch_bounds__(256) void maxpool_kernel(const _Float16* __restrict__ A,
                                                      _Float16* __restrict__ concat,
                                                      int K, int C, int coff) {
    const int total = BATCH * NSAMP * C;
    for (int idx = blockIdx.x * blockDim.x + threadIdx.x; idx < total;
         idx += gridDim.x * blockDim.x) {
        const int q = idx / C;
        const int c = idx % C;
        const _Float16* p = A + (size_t)q * K * C + c;
        float m = -1e30f;
        for (int k = 0; k < K; ++k) m = fmaxf(m, (float)p[(size_t)k * C]);
        concat[(size_t)q * 320 + coff + c] = (_Float16)m;
    }
}

__global__ void zero_kernel(float* __restrict__ p, int n) {
    const int i = blockIdx.x * blockDim.x + threadIdx.x;
    if (i < n) p[i] = 0.0f;
}

// ---------------------------------------------------------------------------
extern "C" void kernel_launch(void* const* d_in, const int* in_sizes, int n_in,
                              void* d_out, int out_size, void* d_ws, size_t ws_size,
                              hipStream_t stream) {
    const float* xyz  = (const float*)d_in[0];
    const float* feat = (const float*)d_in[1];

    // Detect param flattening order: insertion (w,b,gamma,beta) vs sorted (b,beta,gamma,w)
    int WO, BO, GO, TO;
    if (in_sizes[2] == 6 * 32) { WO = 0; BO = 1; GO = 2; TO = 3; }
    else                       { BO = 0; TO = 1; GO = 2; WO = 3; }
    auto P = [&](int layer, int which) -> const float* {
        return (const float*)d_in[2 + layer * 4 + which];
    };

    // Workspace carve-up
    char* ws = (char*)d_ws;
    int*      fps_idx = (int*)ws;                               // 8192 * 4      = 32 KB
    float*    stats   = (float*)(ws + 32768);                   // 1024 * 4      = 4 KB
    _Float16* Wt      = (_Float16*)(ws + 32768 + 4096);         // 512*320*2     = 320 KB
    _Float16* concat  = (_Float16*)(ws + 32768 + 4096 + 327680);// 8192*320*2    = 5 MB
    size_t off = 32768 + 4096 + 327680 + (size_t)8192 * 320 * 2;
    off = (off + 255) & ~(size_t)255;
    const size_t bufBytes = (size_t)1048576 * 128 * 2;          // 256 MB each
    _Float16* bufA = (_Float16*)(ws + off);
    _Float16* bufB = (_Float16*)(ws + off + bufBytes);

    float* new_xyz = (float*)d_out;                              // (8,1024,3)
    float* nf_out  = (float*)d_out + (size_t)BATCH * NSAMP * 3;  // (8,1024,512)

    // 1) FPS -> new_xyz
    fps_kernel<<<BATCH, 256, 0, stream>>>(xyz, new_xyz, fps_idx);

    const int   Ks[3]      = {16, 32, 128};
    const float r2s[3]     = {0.1f * 0.1f, 0.2f * 0.2f, 0.4f * 0.4f};
    const int   mlp[3][3]  = {{32, 32, 64}, {64, 64, 128}, {64, 96, 128}};
    const int   coffs[3]   = {0, 64, 192};

    // 2) Branches
    for (int br = 0; br < 3; ++br) {
        const int K = Ks[br];
        const long long M = (long long)BATCH * NSAMP * K;

        ball_gather_kernel<<<(BATCH * NSAMP) / 4, 128, 0, stream>>>(
            xyz, feat, new_xyz, bufA, K, r2s[br]);

        _Float16* cur = bufA;
        _Float16* nxt = bufB;
        int cin = 6, cin_pad = 32;
        for (int l = 0; l < 3; ++l) {
            const int layer = br * 3 + l;
            const int cout = mlp[br][l];
            prep_w_kernel<<<(cout * cin_pad + 255) / 256, 256, 0, stream>>>(
                P(layer, WO), Wt, cin, cin_pad, cout);
            zero_kernel<<<1, 1024, 0, stream>>>(stats, 2 * cout);
            const long long tiles = (M >> 4) * (long long)(cout >> 5);
            const int blocks = (int)((tiles + 7) / 8);
            gemm_kernel<<<blocks, 256, 0, stream>>>(cur, Wt, P(layer, BO),
                                                    nxt, stats, M, cin_pad, cout);
            bn_relu_kernel<<<2048, 256, 0, stream>>>(nxt, stats, P(layer, GO), P(layer, TO),
                                                     M, cout, nullptr);
            _Float16* t = cur; cur = nxt; nxt = t;
            cin = cout; cin_pad = cout;
        }
        const int C = mlp[br][2];
        maxpool_kernel<<<(BATCH * NSAMP * C + 255) / 256, 256, 0, stream>>>(
            cur, concat, K, C, coffs[br]);
    }

    // 3) Final MLPs: 320 -> 256 -> 512
    {
        const long long M = (long long)BATCH * NSAMP;
        _Float16* cur = concat;
        _Float16* dst[2] = {bufA, bufB};
        const int fouts[2] = {256, 512};
        int cin = 320, cin_pad = 320;
        for (int l = 0; l < 2; ++l) {
            const int layer = 9 + l;
            const int cout = fouts[l];
            prep_w_kernel<<<(cout * cin_pad + 255) / 256, 256, 0, stream>>>(
                P(layer, WO), Wt, cin, cin_pad, cout);
            zero_kernel<<<1, 1024, 0, stream>>>(stats, 2 * cout);
            const long long tiles = (M >> 4) * (long long)(cout >> 5);
            const int blocks = (int)((tiles + 7) / 8);
            gemm_kernel<<<blocks, 256, 0, stream>>>(cur, Wt, P(layer, BO),
                                                    dst[l], stats, M, cin_pad, cout);
            bn_relu_kernel<<<512, 256, 0, stream>>>(dst[l], stats, P(layer, GO), P(layer, TO),
                                                    M, cout, (l == 1) ? nf_out : nullptr);
            cur = dst[l];
            cin = cout; cin_pad = cout;
        }
    }
}
